// MambaBackbone_37890201485395
// MI455X (gfx1250) — matrix operand order repaired and checked by reading, not compiled, not run on hardware
//
#include <hip/hip_runtime.h>

// ---------------- problem constants (from reference) ----------------
#define NDIR   2
#define NLAYER 2
#define BB     4
#define LL     2048
#define DM     512          // d_model
#define DI     1024         // d_inner
#define DSTATE 16
#define DTR    32           // dt_rank
#define DCONV  4
#define NVOCAB 128
#define ROWS   (BB*LL)      // 8192

typedef __attribute__((ext_vector_type(16))) __bf16 v16bf;
typedef __attribute__((ext_vector_type(8)))  float  v8f;

union FragB16 { v16bf v; unsigned u[8]; };
union AccF    { v8f   v; float    f[8]; };

// -------- CDNA5 async global->LDS copy (ASYNCcnt path), inline asm --------
__device__ __forceinline__ void async_load_b128(unsigned lds_addr, const void* gaddr) {
    asm volatile("global_load_async_to_lds_b128 %0, %1, off"
                 :: "v"(lds_addr), "v"(gaddr) : "memory");
}
__device__ __forceinline__ void wait_async0() {
    asm volatile("s_wait_asynccnt 0" ::: "memory");
}

// ---------------- f32 -> bf16 convert ----------------
__global__ __launch_bounds__(256) void f32_to_bf16_kernel(const float* __restrict__ s,
                                                          __bf16* __restrict__ d, int n) {
    int i = blockIdx.x * 256 + threadIdx.x;
    if (i < n) d[i] = (__bf16)s[i];
}

// ---------------- embedding gather (optionally flipped sequence) ----------------
__global__ __launch_bounds__(256) void embed_kernel(const int* __restrict__ ids,
                                                    const float* __restrict__ emb,
                                                    float* __restrict__ h, int flip) {
    int idx = blockIdx.x * 256 + threadIdx.x;            // over ROWS*DM
    if (idx >= ROWS * DM) return;
    int c   = idx & (DM - 1);
    int row = idx >> 9;                                  // DM = 512
    int l   = row & (LL - 1);
    int b   = row >> 11;                                 // LL = 2048
    int lid = flip ? (LL - 1 - l) : l;
    int tok = ids[b * LL + lid];
    h[idx]  = emb[(size_t)tok * DM + c];
}

// ---------------- residual add + RMSNorm -> bf16 ----------------
__global__ __launch_bounds__(128) void fuse_norm_kernel(const float* __restrict__ h,
                                                        float* __restrict__ res,
                                                        const float* __restrict__ w,
                                                        __bf16* __restrict__ outbf, int first) {
    __shared__ float red[128];
    int r = blockIdx.x, t = threadIdx.x;
    float v[4], ss = 0.f;
    #pragma unroll
    for (int i = 0; i < 4; ++i) {
        int c = t + i * 128;
        float x = h[(size_t)r * DM + c];
        if (!first) x += res[(size_t)r * DM + c];
        res[(size_t)r * DM + c] = x;
        v[i] = x; ss += x * x;
    }
    red[t] = ss; __syncthreads();
    for (int o = 64; o > 0; o >>= 1) { if (t < o) red[t] += red[t + o]; __syncthreads(); }
    float rstd = rsqrtf(red[0] / (float)DM + 1e-5f);
    #pragma unroll
    for (int i = 0; i < 4; ++i) {
        int c = t + i * 128;
        outbf[(size_t)r * DM + c] = (__bf16)(v[i] * rstd * w[c]);
    }
}

// ---------------- final RMSNorm -> concat buffer (bf16), bwd dir un-flipped ----------------
__global__ __launch_bounds__(128) void final_norm_cat_kernel(const float* __restrict__ h,
                                                             const float* __restrict__ res,
                                                             const float* __restrict__ w,
                                                             __bf16* __restrict__ cat,
                                                             int colOff, int flip) {
    __shared__ float red[128];
    int r = blockIdx.x, t = threadIdx.x;
    int l = r & (LL - 1), b = r >> 11;
    float v[4], ss = 0.f;
    #pragma unroll
    for (int i = 0; i < 4; ++i) {
        int c = t + i * 128;
        float x = h[(size_t)r * DM + c] + res[(size_t)r * DM + c];
        v[i] = x; ss += x * x;
    }
    red[t] = ss; __syncthreads();
    for (int o = 64; o > 0; o >>= 1) { if (t < o) red[t] += red[t + o]; __syncthreads(); }
    float rstd = rsqrtf(red[0] / (float)DM + 1e-5f);
    int rout = flip ? (b * LL + (LL - 1 - l)) : r;
    #pragma unroll
    for (int i = 0; i < 4; ++i) {
        int c = t + i * 128;
        cat[(size_t)rout * (2 * DM) + colOff + c] = (__bf16)(v[i] * rstd * w[c]);
    }
}

// ---------------- causal depthwise conv (width 4) + SiLU ----------------
__global__ __launch_bounds__(256) void conv_silu_kernel(const float* __restrict__ xz,
                                                        const float* __restrict__ cw,
                                                        const float* __restrict__ cb,
                                                        float* __restrict__ ucv,
                                                        __bf16* __restrict__ ucvb) {
    int idx = blockIdx.x * 256 + threadIdx.x;            // over ROWS*DI
    if (idx >= ROWS * DI) return;
    int ch  = idx & (DI - 1);
    int row = idx >> 10;                                 // DI = 1024
    int l   = row & (LL - 1);
    int b   = row >> 11;
    float acc = cb[ch];
    #pragma unroll
    for (int j = 0; j < DCONV; ++j) {
        int li = l - (DCONV - 1) + j;
        if (li >= 0)
            acc += cw[ch * DCONV + j] * xz[(size_t)(b * LL + li) * (2 * DI) + ch];
    }
    float s = acc / (1.f + __expf(-acc));
    ucv[idx]  = s;
    ucvb[idx] = (__bf16)s;
}

// ---------------- selective scan: single wave32 block, 16-state in VGPRs ----------------
// grid (DI/32, BB); barrier ops are NOPs for single-wave workgroups.
__global__ __launch_bounds__(32) void scan_kernel(const float* __restrict__ u,
                                                  const float* __restrict__ dtp,
                                                  const float* __restrict__ xdbl, // B cols 32..47, C cols 48..63 (ld 64)
                                                  const float* __restrict__ xz,   // z at col DI+ch (ld 2*DI)
                                                  const float* __restrict__ alog,
                                                  const float* __restrict__ Dp,
                                                  __bf16* __restrict__ out) {
    __shared__ float sBC[32];
    int ch = blockIdx.x * 32 + threadIdx.x;
    int b  = blockIdx.y;
    float hs[DSTATE], An[DSTATE];
    #pragma unroll
    for (int n = 0; n < DSTATE; ++n) {
        hs[n] = 0.f;
        An[n] = -__expf(alog[(size_t)ch * DSTATE + n]);
    }
    float dval = Dp[ch];
    for (int t = 0; t < LL; ++t) {
        size_t row = (size_t)b * LL + t;
        float ut = u  [row * DI + ch];
        float dt = dtp[row * DI + ch];
        float zt = xz [row * (2 * DI) + DI + ch];
        sBC[threadIdx.x] = xdbl[row * 64 + 32 + threadIdx.x];
        __syncthreads();
        float du = dt * ut, y = 0.f;
        #pragma unroll
        for (int n = 0; n < DSTATE; ++n) {
            float dA = __expf(dt * An[n]);
            hs[n] = dA * hs[n] + du * sBC[n];
            y += hs[n] * sBC[DSTATE + n];
        }
        y += ut * dval;
        float g = zt / (1.f + __expf(-zt));
        out[row * DI + ch] = (__bf16)(y * g);
        __syncthreads();
    }
}

// ---------------- WMMA bf16 GEMM: C[M,N] = act(A[M,K] * W[N,K]^T + bias) ----------------
// block 256 thr (8 wave32), tile 256x64, K step 32, double-buffered async LDS staging.
// Each wave: 32 rows x 64 cols = 8 v_wmma per K step. grid = (N/64, M/256).
// ACT: 0 none, 1 +bias, 2 softplus(x+bias)

__device__ __forceinline__ void stage_tiles(const __bf16* A, const __bf16* W,
                                            int m0, int n0, int kb, int lda, int ldw,
                                            __bf16* sAbuf, __bf16* sWbuf, int tid) {
    // A tile: 256 x 32 bf16 = 1024 x b128; 4 transfers/thread
    #pragma unroll
    for (int it = 0; it < 4; ++it) {
        int i = tid + it * 256;
        int r = i >> 2, seg = i & 3;
        const __bf16* g = A + (size_t)(m0 + r) * lda + kb + seg * 8;
        unsigned laddr  = (unsigned)(uintptr_t)(sAbuf + r * 32 + seg * 8);
        async_load_b128(laddr, g);
    }
    // W tile: 64 x 32 bf16 = 256 x b128; 1 transfer/thread
    {
        int r = tid >> 2, seg = tid & 3;
        const __bf16* g = W + (size_t)(n0 + r) * ldw + kb + seg * 8;
        unsigned laddr  = (unsigned)(uintptr_t)(sWbuf + r * 32 + seg * 8);
        async_load_b128(laddr, g);
    }
}

template <int ACT>
__global__ __launch_bounds__(256) void gemm_bf16_kernel(const __bf16* __restrict__ A,
                                                        const __bf16* __restrict__ W,
                                                        float* __restrict__ C,
                                                        __bf16* __restrict__ Cbf,
                                                        const float* __restrict__ bias,
                                                        int K, int lda, int ldw, int ldc) {
    __shared__ __bf16 sA[2][256 * 32];   // 2 x 16 KB
    __shared__ __bf16 sW[2][64 * 32];    // 2 x  4 KB
    const int tid  = threadIdx.x;
    const int lane = tid & 31;
    const int wave = tid >> 5;
    const int m0   = blockIdx.y * 256;
    const int n0   = blockIdx.x * 64;

    AccF acc[2][4];
    #pragma unroll
    for (int s = 0; s < 2; ++s)
        #pragma unroll
        for (int j = 0; j < 4; ++j)
            #pragma unroll
            for (int e = 0; e < 8; ++e) acc[s][j].f[e] = 0.f;

    const int mfrag = lane & 15;   // row within A frag / col within B frag
    const int kh    = lane >> 4;   // lane half selects K group

    stage_tiles(A, W, m0, n0, 0, lda, ldw, sA[0], sW[0], tid);
    wait_async0();
    __syncthreads();

    int buf = 0;
    for (int kb = 0; kb < K; kb += 32) {
        int nkb = kb + 32;
        if (nkb < K)   // prefetch next tile into the other buffer (overlaps WMMA)
            stage_tiles(A, W, m0, n0, nkb, lda, ldw, sA[buf ^ 1], sW[buf ^ 1], tid);

        const unsigned* sA32 = (const unsigned*)sA[buf];
        const unsigned* sW32 = (const unsigned*)sW[buf];

        FragB16 a0, a1;
        #pragma unroll
        for (int p = 0; p < 8; ++p) {
            int k0 = ((p < 4) ? 0 : 16) + kh * 8 + ((p & 3) << 1);
            a0.u[p] = sA32[((wave * 32 + mfrag) << 4) + (k0 >> 1)];
            a1.u[p] = sA32[((wave * 32 + 16 + mfrag) << 4) + (k0 >> 1)];
        }
        #pragma unroll
        for (int j = 0; j < 4; ++j) {
            FragB16 bfr;
            #pragma unroll
            for (int p = 0; p < 8; ++p) {
                int k0 = ((p < 4) ? 0 : 16) + kh * 8 + ((p & 3) << 1);
                bfr.u[p] = sW32[((j * 16 + mfrag) << 4) + (k0 >> 1)];
            }
            acc[0][j].v = __builtin_amdgcn_wmma_f32_16x16x32_bf16(
                false, a0.v, false, bfr.v, (short)0, acc[0][j].v, false, false);
            acc[1][j].v = __builtin_amdgcn_wmma_f32_16x16x32_bf16(
                false, a1.v, false, bfr.v, (short)0, acc[1][j].v, false, false);
        }
        wait_async0();
        __syncthreads();
        buf ^= 1;
    }

    // C/D layout: lane&15 = column; VGPR r8 -> row r8 + 8*(lane>>4)
    const int ncol = lane & 15;
    #pragma unroll
    for (int sub = 0; sub < 2; ++sub) {
        #pragma unroll
        for (int j = 0; j < 4; ++j) {
            int col  = n0 + j * 16 + ncol;
            float bv = (ACT != 0) ? bias[col] : 0.f;
            #pragma unroll
            for (int r8 = 0; r8 < 8; ++r8) {
                int row = m0 + wave * 32 + sub * 16 + r8 + 8 * kh;
                float v = acc[sub][j].f[r8];
                if (ACT == 1) v += bv;
                else if (ACT == 2) { v += bv; v = (v > 20.f) ? v : log1pf(__expf(v)); }
                size_t o = (size_t)row * ldc + col;
                C[o] = v;
                if (Cbf) Cbf[o] = (__bf16)v;
            }
        }
    }
}

static void launch_gemm(int act, const __bf16* A, const __bf16* W, float* C, __bf16* Cbf,
                        const float* bias, int M, int N, int K, int lda, int ldw, int ldc,
                        hipStream_t s) {
    dim3 g(N / 64, M / 256), b(256);
    if (act == 0)      gemm_bf16_kernel<0><<<g, b, 0, s>>>(A, W, C, Cbf, bias, K, lda, ldw, ldc);
    else if (act == 1) gemm_bf16_kernel<1><<<g, b, 0, s>>>(A, W, C, Cbf, bias, K, lda, ldw, ldc);
    else               gemm_bf16_kernel<2><<<g, b, 0, s>>>(A, W, C, Cbf, bias, K, lda, ldw, ldc);
}

// ---------------- orchestration ----------------
extern "C" void kernel_launch(void* const* d_in, const int* in_sizes, int n_in,
                              void* d_out, int out_size, void* d_ws, size_t ws_size,
                              hipStream_t stream) {
    (void)in_sizes; (void)n_in; (void)out_size; (void)ws_size;
    const int*   ids       = (const int*)  d_in[0];
    // d_in[1] = padding_mask (all ones, unused by reference math)
    const float* embed_w   = (const float*)d_in[2];
    const float* in_proj_w = (const float*)d_in[3];
    const float* conv_w    = (const float*)d_in[4];
    const float* conv_b    = (const float*)d_in[5];
    const float* x_proj_w  = (const float*)d_in[6];
    const float* dt_proj_w = (const float*)d_in[7];
    const float* dt_proj_b = (const float*)d_in[8];
    const float* A_log     = (const float*)d_in[9];
    const float* Dp        = (const float*)d_in[10];
    const float* out_proj_w= (const float*)d_in[11];
    const float* norm_w    = (const float*)d_in[12];
    const float* norm_f_w  = (const float*)d_in[13];
    const float* proj_w    = (const float*)d_in[14];
    const float* proj_b    = (const float*)d_in[15];
    float* out = (float*)d_out;

    char* base = (char*)d_ws;
    size_t off = 0;
    auto alloc = [&](size_t bytes) -> void* {
        void* p = base + off;
        off += (bytes + 255) & ~(size_t)255;
        return p;
    };

    // bf16 weight mirrors
    __bf16* w_in_bf   = (__bf16*)alloc(sizeof(__bf16) * NDIR * NLAYER * 2 * DI * DM);
    __bf16* w_x_bf    = (__bf16*)alloc(sizeof(__bf16) * NDIR * NLAYER * 64 * DI);
    __bf16* w_dt_bf   = (__bf16*)alloc(sizeof(__bf16) * NDIR * NLAYER * DI * DTR);
    __bf16* w_out_bf  = (__bf16*)alloc(sizeof(__bf16) * NDIR * NLAYER * DM * DI);
    __bf16* w_proj_bf = (__bf16*)alloc(sizeof(__bf16) * DM * 2 * DM);
    // activations
    float*  h     = (float*) alloc(4ull * ROWS * DM);
    float*  res   = (float*) alloc(4ull * ROWS * DM);
    __bf16* hn    = (__bf16*)alloc(2ull * ROWS * DM);
    float*  xz    = (float*) alloc(4ull * ROWS * 2 * DI);
    float*  ucv   = (float*) alloc(4ull * ROWS * DI);
    __bf16* ucvb  = (__bf16*)alloc(2ull * ROWS * DI);
    float*  xdbl  = (float*) alloc(4ull * ROWS * 64);
    __bf16* xdblb = (__bf16*)alloc(2ull * ROWS * 64);
    float*  dtbuf = (float*) alloc(4ull * ROWS * DI);
    __bf16* gate  = (__bf16*)alloc(2ull * ROWS * DI);
    __bf16* cat   = (__bf16*)alloc(2ull * ROWS * 2 * DM);

    auto cvt = [&](const float* s, __bf16* d, int n) {
        f32_to_bf16_kernel<<<(n + 255) / 256, 256, 0, stream>>>(s, d, n);
    };
    cvt(in_proj_w,  w_in_bf,   NDIR * NLAYER * 2 * DI * DM);
    cvt(x_proj_w,   w_x_bf,    NDIR * NLAYER * 64 * DI);
    cvt(dt_proj_w,  w_dt_bf,   NDIR * NLAYER * DI * DTR);
    cvt(out_proj_w, w_out_bf,  NDIR * NLAYER * DM * DI);
    cvt(proj_w,     w_proj_bf, DM * 2 * DM);

    for (int d = 0; d < NDIR; ++d) {
        embed_kernel<<<ROWS * DM / 256, 256, 0, stream>>>(
            ids, embed_w + (size_t)d * NVOCAB * DM, h, d == 1);
        for (int i = 0; i < NLAYER; ++i) {
            int bi = d * NLAYER + i;
            fuse_norm_kernel<<<ROWS, 128, 0, stream>>>(
                h, res, norm_w + (size_t)bi * DM, hn, i == 0);
            // xz = hn @ in_proj_w^T : (8192 x 2048)
            launch_gemm(0, hn, w_in_bf + (size_t)bi * 2 * DI * DM, xz, nullptr, nullptr,
                        ROWS, 2 * DI, DM, DM, DM, 2 * DI, stream);
            conv_silu_kernel<<<ROWS * DI / 256, 256, 0, stream>>>(
                xz, conv_w + (size_t)bi * DI * DCONV, conv_b + (size_t)bi * DI, ucv, ucvb);
            // xdbl = u @ x_proj_w^T : (8192 x 64)  [dt | B | C]
            launch_gemm(0, ucvb, w_x_bf + (size_t)bi * 64 * DI, xdbl, xdblb, nullptr,
                        ROWS, 64, DI, DI, DI, 64, stream);
            // dt = softplus(xdbl[:, :32] @ dt_proj_w^T + dt_b) : (8192 x 1024)
            launch_gemm(2, xdblb, w_dt_bf + (size_t)bi * DI * DTR, dtbuf, nullptr,
                        dt_proj_b + (size_t)bi * DI,
                        ROWS, DI, DTR, 64, DTR, DI, stream);
            scan_kernel<<<dim3(DI / 32, BB), 32, 0, stream>>>(
                ucv, dtbuf, xdbl, xz,
                A_log + (size_t)bi * DI * DSTATE, Dp + (size_t)bi * DI, gate);
            // h = gate @ out_proj_w^T : (8192 x 512)
            launch_gemm(0, gate, w_out_bf + (size_t)bi * DM * DI, h, nullptr, nullptr,
                        ROWS, DM, DI, DI, DI, DM, stream);
        }
        final_norm_cat_kernel<<<ROWS, 128, 0, stream>>>(
            h, res, norm_f_w + (size_t)d * DM, cat, d * DM, d == 1);
    }
    // out = cat @ proj_w^T + proj_b : (8192 x 512)
    launch_gemm(1, cat, w_proj_bf, out, nullptr, proj_b,
                ROWS, DM, 2 * DM, 2 * DM, 2 * DM, DM, stream);
}
